// VLLMAttention_73976516706662
// MI455X (gfx1250) — matrix operand order repaired and checked by reading, not compiled, not run on hardware
//
#include <hip/hip_runtime.h>
#include <hip/hip_bf16.h>

// ---------------------------------------------------------------------------
// CDNA5 (gfx1250) attention block: QKV GEMM -> RMSNorm+RoPE -> causal flash
// attention -> O-proj GEMM.  Matmuls use v_wmma_f32_16x16x32_bf16; GEMM and
// attention K-tiles are staged global->LDS with the Tensor Data Mover
// (tensor_load_to_lds, TENSORcnt), double-buffered in the GEMM.
// ---------------------------------------------------------------------------

typedef __attribute__((ext_vector_type(16))) __bf16 v16bf;
typedef __attribute__((ext_vector_type(8)))  float  v8f;
typedef unsigned int v4u __attribute__((ext_vector_type(4)));
typedef int          v4i __attribute__((ext_vector_type(4)));
typedef int          v8i __attribute__((ext_vector_type(8)));

#define S_LEN   2048
#define HIDDEN  2048
#define HEADS   32
#define HDIM    128
#define QKV_N   12288          // 3 * HEADS * HDIM
#define OH      4096           // HEADS * HDIM
#define ATT_SCALE 0.08838834764831845f   // 128^-0.5

struct alignas(32) BfPack { uint4 lo, hi; };

static __device__ __forceinline__ v16bf pack_frag(uint4 lo, uint4 hi) {
  BfPack t{lo, hi};
  return __builtin_bit_cast(v16bf, t);
}

static __device__ __forceinline__ v8f wmma_bf16(v16bf a, v16bf b, v8f c) {
  // D = A(16x32 bf16) * B(32x16 bf16) + C(16x16 f32)
  return __builtin_amdgcn_wmma_f32_16x16x32_bf16(
      /*neg_a=*/false, a, /*neg_b=*/false, b,
      /*c_mod=*/(short)0, c, /*reuse_a=*/false, /*reuse_b=*/false);
}

// Generic pointer to a __shared__ object keeps the LDS byte offset in the low
// 32 bits (ISA 10.2: LDS aperture address truncates to LDS_ADDR.U32).
static __device__ __forceinline__ unsigned int lds_off(const void* p) {
  return (unsigned int)(uintptr_t)p;
}

// ---------------------------------------------------------------------------
// TDM: issue a 2D tile load global->LDS.  D# per CDNA5 ISA ch.8:
//   group0: [1:0]=count=1, [63:32]=lds_addr, [120:64]=global_addr, [127:126]=2
//   group1: [17:16]=data_size(2 -> 4B), [79:48]=tensor_dim0, [111:80]=tensor_dim1,
//           [127:112]=tile_dim0, [143:128]=tile_dim1, [207:160]=tensor_dim0_stride
// All sizes/strides in 4-byte units.  Issue from ONE wave only (EXEC ignored,
// one DMA per issuing wave); completion via s_wait_tensorcnt.
// ---------------------------------------------------------------------------
static __device__ __forceinline__ void tdm_load_2d(unsigned int lds_byte,
                                                   const void* gptr,
                                                   unsigned int tile_w_dw,
                                                   unsigned int tile_h,
                                                   unsigned int stride_dw,
                                                   unsigned int tensor_w_dw,
                                                   unsigned int tensor_h) {
  unsigned long long ga = (unsigned long long)(uintptr_t)gptr;
  v4u g0;
  g0.x = 1u;                                            // count = 1 (valid D#)
  g0.y = lds_byte;                                      // lds_addr
  g0.z = (unsigned int)ga;                              // global_addr[31:0]
  g0.w = ((unsigned int)(ga >> 32) & 0x01FFFFFFu)       // global_addr[56:32]
         | (2u << 30);                                  // type = 2 ("image")
  v8i g1;
  g1[0] = (int)(2u << 16);                              // data_size = 4B
  g1[1] = (int)((tensor_w_dw & 0xFFFFu) << 16);         // tensor_dim0[15:0]
  g1[2] = (int)(((tensor_w_dw >> 16) & 0xFFFFu)         // tensor_dim0[31:16]
                | ((tensor_h & 0xFFFFu) << 16));        // tensor_dim1[15:0]
  g1[3] = (int)(((tensor_h >> 16) & 0xFFFFu)            // tensor_dim1[31:16]
                | (tile_w_dw << 16));                   // tile_dim0
  g1[4] = (int)(tile_h & 0xFFFFu);                      // tile_dim1 (tile_dim2=0)
  g1[5] = (int)stride_dw;                               // tensor_dim0_stride[31:0]
  g1[6] = 0;                                            // stride[47:32]
  g1[7] = 0;
  v4i z4 = (v4i){0, 0, 0, 0};
#if __clang_major__ >= 23
  v8i z8 = (v8i){0, 0, 0, 0, 0, 0, 0, 0};
  __builtin_amdgcn_tensor_load_to_lds(g0, g1, z4, z4, z8, 0);
#else
  __builtin_amdgcn_tensor_load_to_lds(g0, g1, z4, z4, 0);
#endif
}

// ---------------------------------------------------------------------------
// f32 -> bf16 conversion (memory bound)
// ---------------------------------------------------------------------------
__global__ __launch_bounds__(256) void f32_to_bf16_kernel(const float* __restrict__ src,
                                                          __bf16* __restrict__ dst, int n) {
  int i = blockIdx.x * 256 + threadIdx.x;
  if (i < n) dst[i] = (__bf16)src[i];
}

// ---------------------------------------------------------------------------
// NT GEMM: C[M,N] = A[M,K bf16 row-major] * B[N,K bf16 row-major]^T
// Block tile 128x128, 256 threads (8 waves), K-step 64, TDM-fed double-buffered
// LDS.  Wave w owns rows [16w,16w+16) of the tile, all 128 cols.
// ---------------------------------------------------------------------------
template <bool OUT_BF16>
__global__ __launch_bounds__(256) void gemm_nt_kernel(const __bf16* __restrict__ A,
                                                      const __bf16* __restrict__ B,
                                                      void* __restrict__ Cout,
                                                      int M, int N, int K) {
  __shared__ alignas(32) __bf16 Asm[2][128 * 64];   // 2 x 16KB
  __shared__ alignas(32) __bf16 Bsm[2][128 * 64];   // 2 x 16KB

  const int tid  = threadIdx.x;
  const int wave = tid >> 5;
  const int lane = tid & 31;
  const int m16  = lane & 15;
  const int g    = lane >> 4;

  const int rowBlock = blockIdx.y * 128;
  const int colBlock = blockIdx.x * 128;
  const unsigned int Kdw = (unsigned int)(K >> 1);   // row length in dwords

  v8f acc[8];
#pragma unroll
  for (int j = 0; j < 8; ++j) acc[j] = (v8f){0.f,0.f,0.f,0.f,0.f,0.f,0.f,0.f};

  // prologue: fill buffer 0
  if (tid < 32) {
    tdm_load_2d(lds_off(&Asm[0][0]), A + (size_t)rowBlock * K, 32, 128, Kdw, Kdw, (unsigned)M);
    tdm_load_2d(lds_off(&Bsm[0][0]), B + (size_t)colBlock * K, 32, 128, Kdw, Kdw, (unsigned)N);
  }

  for (int k0 = 0; k0 < K; k0 += 64) {
    const int buf = (k0 >> 6) & 1;
    if (tid < 32) __builtin_amdgcn_s_wait_tensorcnt(0);  // current buf filled
    __syncthreads();                                     // publish fill; prev reads done
    if (k0 + 64 < K && tid < 32) {                       // prefetch next slice
      tdm_load_2d(lds_off(&Asm[buf ^ 1][0]), A + (size_t)rowBlock * K + k0 + 64,
                  32, 128, Kdw, Kdw, (unsigned)M);
      tdm_load_2d(lds_off(&Bsm[buf ^ 1][0]), B + (size_t)colBlock * K + k0 + 64,
                  32, 128, Kdw, Kdw, (unsigned)N);
    }

    const int ar = wave * 16 + m16;
#pragma unroll
    for (int kk = 0; kk < 2; ++kk) {
      uint4 alo = *reinterpret_cast<const uint4*>(&Asm[buf][ar * 64 + kk * 32 + 16 * g]);
      uint4 ahi = *reinterpret_cast<const uint4*>(&Asm[buf][ar * 64 + kk * 32 + 16 * g + 8]);
      v16bf af = pack_frag(alo, ahi);
#pragma unroll
      for (int j = 0; j < 8; ++j) {
        const int br = j * 16 + m16;
        v16bf bf = *reinterpret_cast<const v16bf*>(&Bsm[buf][br * 64 + kk * 32 + 16 * g]);
        acc[j] = wmma_bf16(af, bf, acc[j]);
      }
    }
  }

  // ---- store: C[m][n], m = i + 8*g, n = lane%16 within each 16x16 tile ----
#pragma unroll
  for (int j = 0; j < 8; ++j) {
#pragma unroll
    for (int i = 0; i < 8; ++i) {
      const int row = rowBlock + wave * 16 + i + 8 * g;
      const int col = colBlock + j * 16 + m16;
      if (OUT_BF16)
        ((__bf16*)Cout)[(size_t)row * N + col] = (__bf16)acc[j][i];
      else
        ((float*)Cout)[(size_t)row * N + col] = acc[j][i];
    }
  }
}

// ---------------------------------------------------------------------------
// RMSNorm + RoPE applied in place to the q and k regions of the bf16 QKV
// buffer.  One 128-thread block per (s, head).
// ---------------------------------------------------------------------------
__global__ __launch_bounds__(128) void norm_rope_kernel(__bf16* __restrict__ qkv,
                                                        const int* __restrict__ positions,
                                                        const float* __restrict__ q_w,
                                                        const float* __restrict__ k_w) {
  __shared__ float red[128];
  __shared__ float xbuf[128];

  const int s    = blockIdx.x;
  const int head = blockIdx.y;
  const int d    = threadIdx.x;
  const float pos = (float)positions[s];

  __bf16* base = qkv + (size_t)s * QKV_N + head * HDIM;

  for (int which = 0; which < 2; ++which) {
    __bf16* p = base + which * OH;             // q region then k region
    const float* w = which ? k_w : q_w;

    float x = (float)p[d];
    red[d] = x * x;
    __syncthreads();
#pragma unroll
    for (int off = 64; off >= 1; off >>= 1) {
      if (d < off) red[d] += red[d + off];
      __syncthreads();
    }
    float rms = rsqrtf(red[0] * (1.0f / HDIM) + 1e-5f);
    float xn  = x * rms * w[d];
    xbuf[d] = xn;
    __syncthreads();

    float out;
    const int i = d & 63;
    const float inv  = __powf(10000.0f, -((float)i) * (1.0f / 64.0f));
    const float fr   = pos * inv;
    const float c = __cosf(fr), sn = __sinf(fr);
    if (d < 64) out = xbuf[d] * c - xbuf[d + 64] * sn;
    else        out = xbuf[d] * c + xbuf[i] * sn;
    __syncthreads();                            // protect red/xbuf reuse
    p[d] = (__bf16)out;
  }
}

// ---------------------------------------------------------------------------
// Causal flash attention.  Block = (128 q rows, one head); 8 waves, each wave
// owns 16 q rows.  Key blocks of 32: K-tile staged by TDM, V-tile transposed
// by the waves; online softmax in f32.
// ---------------------------------------------------------------------------
__global__ __launch_bounds__(256) void attention_kernel(const __bf16* __restrict__ qkv,
                                                        __bf16* __restrict__ obuf) {
  __shared__ alignas(32) __bf16 Kt[32 * 128];    // [key][d]  (TDM-filled)
  __shared__ alignas(32) __bf16 Vt[128 * 32];    // [d][key]  (V transposed)
  __shared__ float  Sbuf[8][16 * 32];            // per-wave scores
  __shared__ alignas(32) __bf16 Pbuf[8][16 * 32];// per-wave probs (bf16)
  __shared__ float  mrow[8][16], lrow[8][16], arow[8][16];

  const int tid  = threadIdx.x;
  const int wave = tid >> 5;
  const int lane = tid & 31;
  const int m16  = lane & 15;
  const int g    = lane >> 4;

  const int head     = blockIdx.y;
  const int qBlock   = blockIdx.x * 128;
  const int qRowBase = qBlock + wave * 16;

  // ---- load Q fragments for this wave's 16 rows (4 K-steps over HDIM) ----
  v16bf qf[4];
#pragma unroll
  for (int kk = 0; kk < 4; ++kk) {
    const __bf16* qrow = qkv + (size_t)(qRowBase + m16) * QKV_N + head * HDIM + kk * 32;
    uint4 lo = *reinterpret_cast<const uint4*>(qrow + 16 * g);
    uint4 hi = *reinterpret_cast<const uint4*>(qrow + 16 * g + 8);
    qf[kk] = pack_frag(lo, hi);
  }

  v8f acc[8];
#pragma unroll
  for (int j = 0; j < 8; ++j) acc[j] = (v8f){0.f,0.f,0.f,0.f,0.f,0.f,0.f,0.f};

  if (lane < 16) { mrow[wave][lane] = -INFINITY; lrow[wave][lane] = 0.0f; }

  const int kbEnd = (qBlock + 127) >> 5;         // inclusive causal key-block bound
  for (int kb = 0; kb <= kbEnd; ++kb) {
    __syncthreads();                             // previous-iter LDS reads done
    // ---- TDM: stage K block [32 keys x 128 d] = 32 rows x 64 dwords ----
    if (tid < 32) {
      tdm_load_2d(lds_off(&Kt[0]),
                  qkv + (size_t)(kb * 32) * QKV_N + OH + head * HDIM,
                  64, 32, QKV_N / 2, QKV_N / 2, S_LEN);
    }
    // ---- stage V block transposed: Vt[d][key] (coalesced reads) ----
#pragma unroll
    for (int it = 0; it < 16; ++it) {
      int e   = tid + 256 * it;                  // index over [key][d]
      int key = e >> 7;
      int d   = e & 127;
      Vt[d * 32 + key] =
          qkv[(size_t)(kb * 32 + key) * QKV_N + 2 * OH + head * HDIM + d];
    }
    if (tid < 32) __builtin_amdgcn_s_wait_tensorcnt(0);
    __syncthreads();

    if ((kb << 5) <= qRowBase + 15) {            // wave-uniform causal skip
      // ---- S = Q * K^T  (two 16x16 key sub-tiles) ----
#pragma unroll
      for (int t = 0; t < 2; ++t) {
        v8f s = (v8f){0.f,0.f,0.f,0.f,0.f,0.f,0.f,0.f};
#pragma unroll
        for (int kk = 0; kk < 4; ++kk) {
          v16bf bf = *reinterpret_cast<const v16bf*>(
              &Kt[(t * 16 + m16) * 128 + kk * 32 + 16 * g]);
          s = wmma_bf16(qf[kk], bf, s);
        }
#pragma unroll
        for (int i = 0; i < 8; ++i)
          Sbuf[wave][(i + 8 * g) * 32 + t * 16 + m16] = s[i];
      }

      // ---- online softmax: lane r < 16 owns q row (qRowBase + r) ----
      if (lane < 16) {
        const int q = qRowBase + lane;
        float mo = mrow[wave][lane];
        float lo = lrow[wave][lane];
        float mx = mo;
        float sc[32];
#pragma unroll
        for (int k = 0; k < 32; ++k) {
          float x = Sbuf[wave][lane * 32 + k] * ATT_SCALE;
          if (kb * 32 + k > q) x = -1e30f;       // causal mask
          sc[k] = x;
          mx = fmaxf(mx, x);
        }
        float alpha = __expf(mo - mx);
        float sum = 0.0f;
#pragma unroll
        for (int k = 0; k < 32; ++k) {
          float pv = __expf(sc[k] - mx);
          sum += pv;
          Pbuf[wave][lane * 32 + k] = (__bf16)pv;
        }
        mrow[wave][lane] = mx;
        lrow[wave][lane] = alpha * lo + sum;
        arow[wave][lane] = alpha;
      }
      // per-wave LDS ops are in order; no cross-wave barrier needed here

      // ---- rescale accumulators by alpha of their row ----
      float av[8];
#pragma unroll
      for (int i = 0; i < 8; ++i) av[i] = arow[wave][i + 8 * g];
#pragma unroll
      for (int j = 0; j < 8; ++j)
#pragma unroll
        for (int i = 0; i < 8; ++i) acc[j][i] *= av[i];

      // ---- O += P * V ----
      uint4 plo = *reinterpret_cast<const uint4*>(&Pbuf[wave][m16 * 32 + 16 * g]);
      uint4 phi = *reinterpret_cast<const uint4*>(&Pbuf[wave][m16 * 32 + 16 * g + 8]);
      v16bf pf = pack_frag(plo, phi);
#pragma unroll
      for (int j = 0; j < 8; ++j) {
        v16bf vf = *reinterpret_cast<const v16bf*>(&Vt[(j * 16 + m16) * 32 + 16 * g]);
        acc[j] = wmma_bf16(pf, vf, acc[j]);
      }
    }
  }

  // ---- epilogue: O /= l, store bf16 [s][head*128 + d] ----
  float linv[8];
#pragma unroll
  for (int i = 0; i < 8; ++i) linv[i] = 1.0f / lrow[wave][i + 8 * g];
#pragma unroll
  for (int j = 0; j < 8; ++j)
#pragma unroll
    for (int i = 0; i < 8; ++i) {
      const int row = qRowBase + i + 8 * g;
      const int col = head * HDIM + j * 16 + m16;
      obuf[(size_t)row * OH + col] = (__bf16)(acc[j][i] * linv[i]);
    }
}

// ---------------------------------------------------------------------------
// Host-side launch
// ---------------------------------------------------------------------------
extern "C" void kernel_launch(void* const* d_in, const int* in_sizes, int n_in,
                              void* d_out, int out_size, void* d_ws, size_t ws_size,
                              hipStream_t stream) {
  const float* hidden    = (const float*)d_in[0];   // [1,2048,2048]
  const int*   positions = (const int*)d_in[1];     // [1,2048]
  const float* qkv_w     = (const float*)d_in[2];   // [12288,2048]
  const float* q_norm_w  = (const float*)d_in[3];   // [128]
  const float* k_norm_w  = (const float*)d_in[4];   // [128]
  const float* o_proj_w  = (const float*)d_in[5];   // [2048,4096]
  float* out = (float*)d_out;                       // [1,2048,2048] f32

  char* ws = (char*)d_ws;
  __bf16* X    = (__bf16*)(ws);                            //  8 MB hidden bf16
  __bf16* Wqkv = (__bf16*)(ws + (size_t)8  * 1024 * 1024); // 48 MB qkv_w bf16
  __bf16* Wo   = (__bf16*)(ws + (size_t)56 * 1024 * 1024); // 16 MB o_proj bf16
  __bf16* QKV  = (__bf16*)(ws + (size_t)72 * 1024 * 1024); // 48 MB qkv bf16
  __bf16* Obuf = (__bf16*)(ws + (size_t)120* 1024 * 1024); // 16 MB attn out bf16

  // 1) convert operands to bf16
  {
    int n0 = HIDDEN * S_LEN;            // 4,194,304
    int n1 = QKV_N * HIDDEN;            // 25,165,824
    int n2 = HIDDEN * OH;               //  8,388,608
    f32_to_bf16_kernel<<<n0 / 256, 256, 0, stream>>>(hidden,   X,    n0);
    f32_to_bf16_kernel<<<n1 / 256, 256, 0, stream>>>(qkv_w,    Wqkv, n1);
    f32_to_bf16_kernel<<<n2 / 256, 256, 0, stream>>>(o_proj_w, Wo,   n2);
  }

  // 2) QKV = X * Wqkv^T   (2048 x 12288, K = 2048), bf16 output
  gemm_nt_kernel<true><<<dim3(QKV_N / 128, S_LEN / 128), 256, 0, stream>>>(
      X, Wqkv, QKV, S_LEN, QKV_N, HIDDEN);

  // 3) RMSNorm + RoPE in place on q/k regions of QKV
  norm_rope_kernel<<<dim3(S_LEN, HEADS), 128, 0, stream>>>(
      QKV, positions, q_norm_w, k_norm_w);

  // 4) causal flash attention -> Obuf (bf16, [s][heads*128])
  attention_kernel<<<dim3(S_LEN / 128, HEADS), 256, 0, stream>>>(QKV, Obuf);

  // 5) out = Obuf * Wo^T  (2048 x 2048, K = 4096), f32 output
  gemm_nt_kernel<false><<<dim3(HIDDEN / 128, S_LEN / 128), 256, 0, stream>>>(
      Obuf, Wo, out, S_LEN, HIDDEN, OH);
}
